// RegionLoss_79757542687148
// MI455X (gfx1250) — compile-verified, matrix operand order checked by reading, not run on hardware
//
#include <hip/hip_runtime.h>
#include <hip/hip_bf16.h>

typedef __attribute__((ext_vector_type(16))) _Float16 v16h;
typedef __attribute__((ext_vector_type(8)))  float    v8f;

#define NB 16
#define NT 50
#define NA 5
#define NC 7
#define NH 152
#define NW 152
#define NHW (NH*NW)
#define CELLS (NA*NHW)
#define CHUNKS 64
#define CPC ((CELLS + CHUNKS - 1)/CHUNKS)   // 1805 cells per block
#define TGF 24                               // floats per precomputed target
#define THREADS 256
#define IGNORE_T 0.6f

__device__ __forceinline__ float sigm(float v) { return 1.f / (1.f + __expf(-v)); }

// ---------------------------------------------------------------------------
// Kernel 1: per-target precompute (800 targets) + zero accumulators.
// tg layout per target (24 floats):
//  0:gx 1:gy 2:gw 3:gl 4:gi 5:gj 6:best_n 7:valid 8:valid_prefix
//  9:tx 10:ty 11:tw 12:tl 13:tim 14:tre 15:label 16..20:anch_iou[5] 21..23:pad
// ---------------------------------------------------------------------------
__global__ void rl_prep(const float* __restrict__ target,
                        const float* __restrict__ anchors,
                        float* __restrict__ tg, float* __restrict__ acc) {
    int idx = blockIdx.x * blockDim.x + threadIdx.x;
    if (idx < 16) acc[idx] = 0.f;
    if (idx >= NB * NT) return;
    int b = idx / NT, t = idx - b * NT;
    const float* T = target + (size_t)(b * NT + t) * 7;
    float lab = T[0], cx = T[1], cy = T[2], w = T[3], l = T[4], im = T[5], re = T[6];
    float gx = cx * NW, gy = cy * NH, gw = w * NW, gl = l * NH;
    float rowsum = lab + cx + cy + w + l + im + re;
    float valid = (rowsum != 0.f) ? 1.f : 0.f;
    int vp = 1;
    for (int tt = 0; tt <= t; ++tt)
        vp &= (target[(size_t)(b * NT + tt) * 7 + 1] != 0.f) ? 1 : 0;
    int gi = (int)gx; gi = gi < 0 ? 0 : (gi > NW - 1 ? NW - 1 : gi);
    int gj = (int)gy; gj = gj < 0 ? 0 : (gj > NH - 1 ? NH - 1 : gj);
    float a_gt = (gw + 1.f) * (gl + 1.f);
    float best = -1.f; int bn = 0;
    float ious[NA];
    for (int a = 0; a < NA; ++a) {
        float aw = anchors[2 * a], ah = anchors[2 * a + 1];
        float iw = fminf(gw, aw) + 1.f; iw = iw < 0.f ? 0.f : iw;
        float ih = fminf(gl, ah) + 1.f; ih = ih < 0.f ? 0.f : ih;
        float inter = iw * ih;
        float a_an = (aw + 1.f) * (ah + 1.f);
        float iou = inter / (a_gt + a_an - inter + 1e-16f);
        ious[a] = iou;
        if (iou > best) { best = iou; bn = a; }   // strict > keeps first-argmax
    }
    float* o = tg + (size_t)idx * TGF;
    o[0] = gx; o[1] = gy; o[2] = gw; o[3] = gl;
    o[4] = (float)gi; o[5] = (float)gj; o[6] = (float)bn;
    o[7] = valid; o[8] = (float)vp;
    o[9] = gx - (float)gi; o[10] = gy - (float)gj;
    o[11] = __logf(gw / anchors[2 * bn] + 1e-16f);
    o[12] = __logf(gl / anchors[2 * bn + 1] + 1e-16f);
    o[13] = im; o[14] = re; o[15] = lab;
    for (int a = 0; a < NA; ++a) o[16 + a] = ious[a];
    o[21] = 0.f; o[22] = 0.f; o[23] = 0.f;
}

// ---------------------------------------------------------------------------
// Kernel 2: fused per-cell loss pass. One batch per block (targets in LDS),
// 11 partial sums per thread, reduced per-block with V_WMMA_F32_16X16X32_F16
// (B = ones -> column sums), then float atomics to global accumulators.
// acc: 0:sum(m) 1..4:sx,sy,sw,sh 5:sim 6:sre 7:sum_cf 8:bce*cf 9:bce*m 10:ce*m
// ---------------------------------------------------------------------------
__global__ __launch_bounds__(THREADS) void rl_main(const float* __restrict__ x,
                                                   const float* __restrict__ anchors,
                                                   const float* __restrict__ tg,
                                                   float* __restrict__ acc) {
    __shared__ float ltg[NT * TGF];
    __shared__ float red[THREADS * 17];   // stride 17 to dodge bank conflicts
    const int b = blockIdx.x / CHUNKS;
    const int chunk = blockIdx.x - b * CHUNKS;
    for (int q = threadIdx.x; q < NT * TGF; q += THREADS)
        ltg[q] = tg[(size_t)b * NT * TGF + q];
    __syncthreads();

    float p[11];
#pragma unroll
    for (int q = 0; q < 11; ++q) p[q] = 0.f;

    const int start = chunk * CPC;
    const int end = (start + CPC < CELLS) ? start + CPC : CELLS;
    for (int cell = start + threadIdx.x; cell < end; cell += THREADS) {
        int a = cell / NHW;
        int r = cell - a * NHW;
        int j = r / NW, i = r - j * NW;
        const float* xb = x + ((size_t)(b * NA + a) * 14) * NHW + r;
        __builtin_prefetch(xb + THREADS, 0, 0);  // next cell this thread touches
        float v[14];
#pragma unroll
        for (int c = 0; c < 14; ++c) v[c] = xb[(size_t)c * NHW];

        float px = sigm(v[0]), py = sigm(v[1]);
        float pw = v[2], pl = v[3], pim = v[4], pre = v[5];
        float pconf = sigm(v[6]);
        float aw = anchors[2 * a], ah = anchors[2 * a + 1];
        float bx = px + (float)i, by = py + (float)j;
        float bw = __expf(pw) * aw, bh = __expf(pl) * ah;

        float maxIoU = 0.f;
        bool anyIgn = false;
        int lastT = -1;
        for (int t = 0; t < NT; ++t) {
            const float* g = &ltg[t * TGF];
            if (g[8] != 0.f) {  // valid_prefix: IoU-ignore scan
                float gx = g[0], gy = g[1], gw = g[2], gl = g[3];
                float mx = fminf(bx - bw * 0.5f, gx - gw * 0.5f);
                float Mx = fmaxf(bx + bw * 0.5f, gx + gw * 0.5f);
                float my = fminf(by - bh * 0.5f, gy - gl * 0.5f);
                float My = fmaxf(by + bh * 0.5f, gy + gl * 0.5f);
                float cw = bw + gw - (Mx - mx);
                float ch = bh + gl - (My - my);
                float carea = (cw <= 0.f || ch <= 0.f) ? 0.f : cw * ch;
                float uarea = bw * bh + gw * gl - carea;
                maxIoU = fmaxf(maxIoU, carea / uarea);
            }
            if (g[7] != 0.f && (int)g[4] == i && (int)g[5] == j) {  // valid, cell match
                if (g[16 + a] > IGNORE_T) anyIgn = true;
                if ((int)g[6] == a) lastT = t;                      // last write wins
            }
        }
        float m = (lastT >= 0) ? 1.f : 0.f;
        float conf = (lastT >= 0) ? 1.f
                                  : (anyIgn ? 0.f : (maxIoU <= IGNORE_T ? 1.f : 0.f));
        float pc = fminf(fmaxf(pconf, 1e-12f), 1.f - 1e-12f);
        float bce = -(m * __logf(pc) + (1.f - m) * __logf(1.f - pc));
        float cf = conf * (1.f - m);
        p[0] += m;
        p[7] += cf;
        p[8] += bce * cf;
        p[9] += bce * m;
        if (lastT >= 0) {
            const float* g = &ltg[lastT * TGF];
            float dx = px - g[9], dy = py - g[10], dw = pw - g[11], dh = pl - g[12];
            float di = pim - g[13], dr = pre - g[14];
            p[1] += dx * dx; p[2] += dy * dy; p[3] += dw * dw; p[4] += dh * dh;
            p[5] += di * di; p[6] += dr * dr;
            // ce = -log_softmax(softmax(logits))[label]   (reference quirk preserved)
            float mm = v[7];
#pragma unroll
            for (int k = 8; k < 14; ++k) mm = fmaxf(mm, v[k]);
            float s[NC]; float se = 0.f;
#pragma unroll
            for (int k = 0; k < NC; ++k) { s[k] = __expf(v[7 + k] - mm); se += s[k]; }
            float inv = 1.f / se;
            float mm2 = -1.f;
#pragma unroll
            for (int k = 0; k < NC; ++k) { s[k] *= inv; mm2 = fmaxf(mm2, s[k]); }
            float se2 = 0.f;
#pragma unroll
            for (int k = 0; k < NC; ++k) se2 += __expf(s[k] - mm2);
            float lse2 = mm2 + __logf(se2);
            int lab = (int)g[15];
            if (lab >= 0 && lab < NC) p[10] += -(s[lab] - lse2);
        }
    }

    // ---- block reduction: 256x16 table -> column sums via WMMA ----
    const int tid = threadIdx.x;
#pragma unroll
    for (int q = 0; q < 11; ++q) red[tid * 17 + q] = p[q];
#pragma unroll
    for (int q = 11; q < 17; ++q) red[tid * 17 + q] = 0.f;
    __syncthreads();

    int wave = __builtin_amdgcn_readfirstlane((int)(threadIdx.x >> 5));
    if (wave == 0) {                 // uniform branch: EXEC all-1s inside
        v16h ones;
#pragma unroll
        for (int q = 0; q < 16; ++q) ones[q] = (_Float16)1.0f;
        v8f c = {};
        const int col = tid & 15;
        const bool hi = tid >= 16;
#pragma unroll
        for (int ch = 0; ch < 8; ++ch) {     // 8 chunks of 32 rows
            v16h av;
#pragma unroll
            for (int e = 0; e < 16; ++e) {
                int k = hi ? (e < 8 ? e + 8 : e + 16)    // K = 8..15, 24..31
                           : (e < 8 ? e     : e + 8);    // K = 0..7, 16..23
                av[e] = (_Float16)red[(ch * 32 + k) * 17 + col];
            }
            // D[m][n] = sum_k A[m][k]*1 + C  -> per-column running sum
            c = __builtin_amdgcn_wmma_f32_16x16x32_f16(
                    false, av, false, ones, (short)0, c, false, false);
        }
        if (tid == 0) {              // lane0 VGPR q holds D[M=q][N=0] = colSum[q]
#pragma unroll
            for (int q = 0; q < 8; ++q) atomicAdd(&acc[q], c[q]);
        } else if (tid == 16) {      // lane16 VGPR q holds D[M=8+q][N=0]
#pragma unroll
            for (int q = 0; q < 3; ++q) atomicAdd(&acc[8 + q], c[q]);
        }
    }
}

// ---------------------------------------------------------------------------
// Kernel 3: combine accumulators into the scalar loss.
// ---------------------------------------------------------------------------
__global__ void rl_final(const float* __restrict__ acc, float* __restrict__ out) {
    float nM = fmaxf(acc[0], 1.f);
    float scf = fmaxf(acc[7], 1.f);
    float loss = (acc[1] + acc[2] + acc[3] + acc[4] + acc[5] + acc[6]) / nM
               + acc[8] / scf + acc[9] / nM
               + (1.f / (float)NB) * acc[10] / nM;
    out[0] = loss;
}

extern "C" void kernel_launch(void* const* d_in, const int* in_sizes, int n_in,
                              void* d_out, int out_size, void* d_ws, size_t ws_size,
                              hipStream_t stream) {
    const float* x       = (const float*)d_in[0];
    const float* target  = (const float*)d_in[1];
    const float* anchors = (const float*)d_in[2];
    float* out = (float*)d_out;
    float* tg  = (float*)d_ws;                 // NB*NT*TGF floats (~76.8 KB)
    float* acc = tg + (size_t)NB * NT * TGF;   // 16 floats

    rl_prep<<<(NB * NT + 255) / 256, 256, 0, stream>>>(target, anchors, tg, acc);
    rl_main<<<NB * CHUNKS, THREADS, 0, stream>>>(x, anchors, tg, acc);
    rl_final<<<1, 1, 0, stream>>>(acc, out);
}
